// aggregator_27633819583079
// MI455X (gfx1250) — matrix operand order repaired and checked by reading, not compiled, not run on hardware
//
#include <hip/hip_runtime.h>
#include <hip/hip_bf16.h>

typedef _Float16 v16h __attribute__((ext_vector_type(16)));
typedef float    v8f  __attribute__((ext_vector_type(8)));

#define EMBED_DIM 64
#define MAX_NEIGH 50
#define ROWS      64          // padded neighbor rows (L=50 -> 64)
#define XSTR      136         // f16 stride for X / W1t (128 + 8 pad)
#define HSTR      72          // f16 stride for H / W2t (64 + 8 pad)
#define H2STR     65          // f32 stride for H2

// shared memory byte offsets (all 16B aligned)
#define OFF_NF    0                      // float  [64][64]        16384
#define OFF_X     16384                  // f16    [64][136]       17408 (reused as H2 f32 [64][65])
#define OFF_BUF1  33792                  // f16 W1t[64][136] 17408 | H f16 [64][72] 9216 + W2t f16 [64][72] 9216
#define OFF_NODE  52224                  // f16    [64]            128
#define OFF_W3    52352                  // float  [64]            256
#define OFF_B1    52608                  // float  [64]            256
#define OFF_B2    52864                  // float  [64]            256
#define OFF_LOG   53120                  // float  [64]            256
#define OFF_ATT   53376                  // float  [64]            256
#define SMEM_BYTES 53632

__device__ __forceinline__ v16h ldfrag(const _Float16* p0, const _Float16* p1) {
  union { v16h v; uint4 q[2]; } u;
  u.q[0] = *(const uint4*)p0;
  u.q[1] = *(const uint4*)p1;
  return u.v;
}

__global__ __launch_bounds__(128)
void graphrec_attn_kernel(const float* __restrict__ u_table,
                          const float* __restrict__ i_table,
                          const float* __restrict__ W1,
                          const float* __restrict__ b1,
                          const float* __restrict__ W2,
                          const float* __restrict__ b2,
                          const float* __restrict__ W3,
                          const float* __restrict__ b3,
                          const int*   __restrict__ nodes,
                          const int*   __restrict__ neighbors,
                          const int*   __restrict__ lengths,
                          float*       __restrict__ out) {
  __shared__ __align__(16) char smem[SMEM_BYTES];
  float*    NF   = (float*)   (smem + OFF_NF);
  _Float16* X    = (_Float16*)(smem + OFF_X);
  float*    H2   = (float*)   (smem + OFF_X);     // aliases X (X dead by then)
  _Float16* W1T  = (_Float16*)(smem + OFF_BUF1);
  _Float16* H    = (_Float16*)(smem + OFF_BUF1);  // aliases W1T (after barrier)
  _Float16* W2T  = (_Float16*)(smem + OFF_BUF1 + ROWS * HSTR * 2);
  _Float16* NODE = (_Float16*)(smem + OFF_NODE);
  float*    W3S  = (float*)   (smem + OFF_W3);
  float*    B1S  = (float*)   (smem + OFF_B1);
  float*    B2S  = (float*)   (smem + OFF_B2);
  float*    LOG  = (float*)   (smem + OFF_LOG);
  float*    ATT  = (float*)   (smem + OFF_ATT);

  const int b   = blockIdx.x;
  const int tid = threadIdx.x;
  const int len = lengths[b];
  const int nodeIdx = nodes[b];

  // ---- Phase 0a: small per-column loads (node row, w3, biases) ----
  if (tid < 64) {
    NODE[tid] = (_Float16)u_table[(size_t)nodeIdx * EMBED_DIM + tid];
    W3S[tid]  = W3[tid];
    B1S[tid]  = b1[tid];
    B2S[tid]  = b2[tid];
  }
  __syncthreads();

  // ---- Phase 0b: gather neighbors -> NF (f32) + X (f16), transpose W1 -> W1T ----
  {
    const int l = tid >> 1;         // 0..63
    const int h = tid & 1;          // half-row: cols [h*32, h*32+32)
    if (l < len) {
      const int nidx = neighbors[b * MAX_NEIGH + l];
      const float4* src = (const float4*)(i_table + (size_t)nidx * EMBED_DIM + h * 32);
      float4*       nfd = (float4*)(NF + l * EMBED_DIM + h * 32);
      _Float16*     xd  = X + l * XSTR + h * 32;
#pragma unroll
      for (int q = 0; q < 8; ++q) {
        float4 f = src[q];
        nfd[q] = f;
        xd[q * 4 + 0] = (_Float16)f.x;
        xd[q * 4 + 1] = (_Float16)f.y;
        xd[q * 4 + 2] = (_Float16)f.z;
        xd[q * 4 + 3] = (_Float16)f.w;
      }
      // broadcast node features into cols [64,128)
      const uint4* ns = (const uint4*)(NODE + h * 32);
      uint4*       nd = (uint4*)(X + l * XSTR + 64 + h * 32);
#pragma unroll
      for (int q = 0; q < 4; ++q) nd[q] = ns[q];
    } else {
      // zero padded / masked rows
      uint4 z; z.x = z.y = z.z = z.w = 0u;
      uint4* xd = (uint4*)(X + l * XSTR + h * 64);
#pragma unroll
      for (int q = 0; q < 8; ++q) xd[q] = z;
    }
    // W1 [128,64] row-major -> W1T[n][k] (f16). thread tid handles k = tid.
    const float4* wr = (const float4*)(W1 + tid * EMBED_DIM);
#pragma unroll 4
    for (int n4 = 0; n4 < 16; ++n4) {
      float4 w = wr[n4];
      W1T[(n4 * 4 + 0) * XSTR + tid] = (_Float16)w.x;
      W1T[(n4 * 4 + 1) * XSTR + tid] = (_Float16)w.y;
      W1T[(n4 * 4 + 2) * XSTR + tid] = (_Float16)w.z;
      W1T[(n4 * 4 + 3) * XSTR + tid] = (_Float16)w.w;
    }
  }
  __syncthreads();

  // ---- WMMA lane geometry ----
  const int wv    = tid >> 5;        // wave id: M-tile = wv*16
  const int lane  = tid & 31;
  const int lm    = lane & 15;
  const int lh    = lane >> 4;
  const int arow  = wv * 16 + lm;    // A-fragment row
  const int khalf = lh * 8;          // A: K sub-offset
  const int koff  = lh * 8;          // B: K sub-offset

  // ---- Phase 1: H = relu(X @ W1 + b1), X:[64,128] f16, 16 WMMAs/wave ----
  v8f acc[4];
#pragma unroll
  for (int nt = 0; nt < 4; ++nt) acc[nt] = (v8f){0.f,0.f,0.f,0.f,0.f,0.f,0.f,0.f};
#pragma unroll
  for (int kk = 0; kk < 4; ++kk) {
    const int kb = kk * 32;
    const _Float16* ap = X + arow * XSTR + kb + khalf;
    v16h a = ldfrag(ap, ap + 16);
#pragma unroll
    for (int nt = 0; nt < 4; ++nt) {
      const _Float16* bp = W1T + (nt * 16 + lm) * XSTR + kb + koff;
      v16h bf = ldfrag(bp, bp + 16);
      acc[nt] = __builtin_amdgcn_wmma_f32_16x16x32_f16(
          false, a, false, bf, (short)0, acc[nt], false, false);
    }
  }
  __syncthreads();   // everyone done reading W1T before it becomes H / W2T

  // ---- Phase 1b: store H (relu, f16) into recycled region; load W2T ----
#pragma unroll
  for (int nt = 0; nt < 4; ++nt) {
    const int col = nt * 16 + lm;
    const float bias = B1S[col];
#pragma unroll
    for (int j = 0; j < 8; ++j) {
      const int row = wv * 16 + lh * 8 + j;
      float v = acc[nt][j] + bias;
      H[row * HSTR + col] = (_Float16)(v > 0.f ? v : 0.f);
    }
  }
  if (tid < 64) {
    const float4* wr = (const float4*)(W2 + tid * EMBED_DIM);
#pragma unroll 4
    for (int n4 = 0; n4 < 16; ++n4) {
      float4 w = wr[n4];
      W2T[(n4 * 4 + 0) * HSTR + tid] = (_Float16)w.x;
      W2T[(n4 * 4 + 1) * HSTR + tid] = (_Float16)w.y;
      W2T[(n4 * 4 + 2) * HSTR + tid] = (_Float16)w.z;
      W2T[(n4 * 4 + 3) * HSTR + tid] = (_Float16)w.w;
    }
  }
  __syncthreads();

  // ---- Phase 2: H2 = relu(H @ W2 + b2), 8 WMMAs/wave ----
  v8f acc2[4];
#pragma unroll
  for (int nt = 0; nt < 4; ++nt) acc2[nt] = (v8f){0.f,0.f,0.f,0.f,0.f,0.f,0.f,0.f};
#pragma unroll
  for (int kk = 0; kk < 2; ++kk) {
    const int kb = kk * 32;
    const _Float16* ap = H + arow * HSTR + kb + khalf;
    v16h a = ldfrag(ap, ap + 16);
#pragma unroll
    for (int nt = 0; nt < 4; ++nt) {
      const _Float16* bp = W2T + (nt * 16 + lm) * HSTR + kb + koff;
      v16h bf = ldfrag(bp, bp + 16);
      acc2[nt] = __builtin_amdgcn_wmma_f32_16x16x32_f16(
          false, a, false, bf, (short)0, acc2[nt], false, false);
    }
  }
  // store H2 (f32) into the X region (disjoint from H/W2T being read)
#pragma unroll
  for (int nt = 0; nt < 4; ++nt) {
    const int col = nt * 16 + lm;
    const float bias = B2S[col];
#pragma unroll
    for (int j = 0; j < 8; ++j) {
      const int row = wv * 16 + lh * 8 + j;
      float v = acc2[nt][j] + bias;
      H2[row * H2STR + col] = (v > 0.f ? v : 0.f);
    }
  }
  __syncthreads();

  // ---- Phase 3: logits[r] = H2[r] . w3 + b3 ----
  if (tid < 64) {
    float acc3 = b3[0];
    const float* hr = H2 + tid * H2STR;
#pragma unroll 8
    for (int k = 0; k < EMBED_DIM; ++k) acc3 = fmaf(hr[k], W3S[k], acc3);
    LOG[tid] = acc3;
  }
  __syncthreads();

  // ---- Phase 4: masked softmax + weighted sum of f32 neighbors ----
  float maxv = -3.402823466e38f;
  for (int l = 0; l < len; ++l) maxv = fmaxf(maxv, LOG[l]);
  if (tid < len) ATT[tid] = __expf(LOG[tid] - maxv);
  __syncthreads();
  if (tid < 64) {
    float denom = 0.f;
    for (int l = 0; l < len; ++l) denom += ATT[l];
    const float inv = 1.f / denom;
    float accO = 0.f;
    for (int l = 0; l < len; ++l) accO = fmaf(ATT[l], NF[l * EMBED_DIM + tid], accO);
    out[(size_t)b * EMBED_DIM + tid] = accO * inv;
  }
}

extern "C" void kernel_launch(void* const* d_in, const int* in_sizes, int n_in,
                              void* d_out, int out_size, void* d_ws, size_t ws_size,
                              hipStream_t stream) {
  const float* u_table   = (const float*)d_in[0];
  const float* i_table   = (const float*)d_in[1];
  const float* W1        = (const float*)d_in[2];
  const float* b1        = (const float*)d_in[3];
  const float* W2        = (const float*)d_in[4];
  const float* b2        = (const float*)d_in[5];
  const float* W3        = (const float*)d_in[6];
  const float* b3        = (const float*)d_in[7];
  const int*   nodes     = (const int*)d_in[8];
  const int*   neighbors = (const int*)d_in[9];
  const int*   lengths   = (const int*)d_in[10];
  float*       out       = (float*)d_out;

  const int B = in_sizes[8];   // number of batch nodes
  graphrec_attn_kernel<<<dim3(B), dim3(128), 0, stream>>>(
      u_table, i_table, W1, b1, W2, b2, W3, b3, nodes, neighbors, lengths, out);
}
